// SelfAttention1_68917045231658
// MI455X (gfx1250) — matrix operand order repaired and checked by reading, not compiled, not run on hardware
//
#include <hip/hip_runtime.h>
#include <hip/hip_bf16.h>

typedef __attribute__((ext_vector_type(16))) __bf16 bf16x16;
typedef __attribute__((ext_vector_type(8)))  float  f32x8;

#define NPTS   16384   // B*N
#define CDIM   128
#define KNN    16

// ---------------- WMMA helpers (v_wmma_f32_16x16x32_bf16) ----------------
// A fragment: 16x32 bf16 tile from LDS (row-major 16x128).
// lane m = lane&15 ; elems 0..7 -> K = k0 + (lane>=16?8:0) + e
//                    elems 8..15 -> K = k0 + 16 + (lane>=16?8:0) + e
__device__ __forceinline__ bf16x16 frag_a_lds(const __bf16* X, int lane, int k0) {
  const int m    = lane & 15;
  const int kofs = k0 + ((lane & 16) ? 8 : 0);
  const __bf16* p = X + m * CDIM + kofs;
  bf16x16 a;
#pragma unroll
  for (int e = 0; e < 8; ++e) { a[e] = p[e]; a[8 + e] = p[16 + e]; }
  return a;
}

// B fragment: B[k][n] = W[n][k], W row-major 128x128 bf16 (LDS-staged).
// lane col n = n0 + (lane&15) ; elems 0..7 -> K = k0 + (lane>=16?8:0) + e
//                               elems 8..15 -> K = k0 + 16 + (lane>=16?8:0) + e
__device__ __forceinline__ bf16x16 frag_b_w(const __bf16* W, int lane, int n0, int k0) {
  const int n    = n0 + (lane & 15);
  const int kofs = k0 + ((lane & 16) ? 8 : 0);
  const __bf16* p = W + n * CDIM + kofs;
  bf16x16 b;
#pragma unroll
  for (int e = 0; e < 8; ++e) { b[e] = p[e]; b[8 + e] = p[16 + e]; }
  return b;
}

__device__ __forceinline__ void zero_acc(f32x8* acc) {
#pragma unroll
  for (int nb = 0; nb < 8; ++nb)
#pragma unroll
    for (int j = 0; j < 8; ++j) acc[nb][j] = 0.0f;
}

// Y(16x128) += X(16x128, LDS bf16) @ W^T (W: 128x128 row-major bf16, LDS)
// B-fragment loads pipelined 2 tiles ahead.
__device__ __forceinline__ void gemm16x128x128(const __bf16* Xlds, const __bf16* W,
                                               int lane, f32x8* acc) {
  bf16x16 bq[3];
  bq[0] = frag_b_w(W, lane, 0 * 16, 0);
  bq[1] = frag_b_w(W, lane, 1 * 16, 0);
#pragma unroll
  for (int kb = 0; kb < 4; ++kb) {
    bf16x16 afr = frag_a_lds(Xlds, lane, kb * 32);
#pragma unroll
    for (int nb = 0; nb < 8; ++nb) {
      const int lin = kb * 8 + nb;
      if (lin + 2 < 32) {
        const int nxt = lin + 2;
        bq[nxt % 3] = frag_b_w(W, lane, (nxt & 7) * 16, (nxt >> 3) * 32);
      }
      acc[nb] = __builtin_amdgcn_wmma_f32_16x16x32_bf16(
          false, afr, false, bq[lin % 3], (short)0, acc[nb], false, false);
    }
  }
}

// copy one 128x128 bf16 weight (32KB) from global into LDS, b128-vectorized
__device__ __forceinline__ void stage_w(__bf16* dst, const __bf16* src,
                                        int tid, int nthreads) {
  const uint4* s = (const uint4*)src;
  uint4* d = (uint4*)dst;
  for (int i = tid; i < 2048; i += nthreads) d[i] = s[i];
}

// ---------------- small utility kernels ----------------
__global__ void k_init(float* st) {
  for (int i = threadIdx.x; i < 1024; i += 256) st[i] = 0.0f;
}

struct WPtrs { const float* src[8]; };

__global__ void k_cvt(WPtrs wp, __bf16* dst) {
  int i = blockIdx.x * 256 + threadIdx.x;           // 8 * 16384
  int mid = i >> 14, off = i & 16383;
  dst[i] = (__bf16)wp.src[mid][off];
}

__global__ void k_stats(const float* ssum, const float* ssq,
                        const float* g, const float* bta,
                        float* scale, float* bias) {
  int c = threadIdx.x;                              // 128 threads
  const float invN = 1.0f / (float)NPTS;
  float m = ssum[c] * invN;
  float v = ssq[c] * invN - m * m;
  float sc = g[c] * rsqrtf(v + 1e-5f);
  scale[c] = sc;
  bias[c]  = bta[c] - m * sc;
}

// ---------------- GEMM + BN-stat kernel:  Y = X @ W^T + b, accumulate sum/sumsq ----------------
__global__ void k_conv_stats(const float* __restrict__ X, const __bf16* __restrict__ W,
                             const float* __restrict__ bias,
                             float* __restrict__ Y, float* ssum, float* ssq) {
  __shared__ __bf16 lx[4][16 * CDIM];               // 16 KB
  __shared__ __bf16 s_w[CDIM * CDIM];               // 32 KB weight cache
  const int wave = threadIdx.x >> 5, lane = threadIdx.x & 31;
  const int tile = blockIdx.x * 4 + wave;           // 1024 tiles of 16 rows
  const int r0 = tile * 16;
  stage_w(s_w, W, threadIdx.x, 128);
  __bf16* xl = lx[wave];
  for (int i = lane; i < 16 * CDIM; i += 32) xl[i] = (__bf16)X[r0 * CDIM + i];
  __syncthreads();

  f32x8 acc[8]; zero_acc(acc);
  gemm16x128x128(xl, s_w, lane, acc);

  const int hi = (lane & 16) ? 8 : 0;
#pragma unroll
  for (int nb = 0; nb < 8; ++nb) {
    const int n = nb * 16 + (lane & 15);
    const float bv = bias[n];
    float s = 0.f, s2 = 0.f;
#pragma unroll
    for (int j = 0; j < 8; ++j) {
      float v = acc[nb][j] + bv;
      Y[(r0 + hi + j) * CDIM + n] = v;
      s += v; s2 += v * v;
    }
    // lanes l and l^16 share channel n (rows 0-7 vs 8-15): combine, then one atomic
    s  += __shfl_xor(s, 16, 32);
    s2 += __shfl_xor(s2, 16, 32);
    if (!(lane & 16)) {
      atomicAdd(&ssum[n], s);
      atomicAdd(&ssq[n], s2);
    }
  }
}

// ---------------- x = relu(bn2(y2)); q/k/v = x @ W{q,k,v}^T ----------------
__global__ void k_qkv(const float* __restrict__ Y2, const float* __restrict__ scale,
                      const float* __restrict__ bias,
                      const __bf16* __restrict__ Wq, const __bf16* __restrict__ Wk,
                      const __bf16* __restrict__ Wv,
                      float* __restrict__ Q, float* __restrict__ Kf, float* __restrict__ Vf) {
  __shared__ __bf16 lx[4][16 * CDIM];               // 16 KB
  __shared__ __bf16 s_w[CDIM * CDIM];               // 32 KB, cycled Wq->Wk->Wv
  const int wave = threadIdx.x >> 5, lane = threadIdx.x & 31;
  const int tile = blockIdx.x * 4 + wave;
  const int r0 = tile * 16;
  stage_w(s_w, Wq, threadIdx.x, 128);
  __bf16* xl = lx[wave];
  for (int i = lane; i < 16 * CDIM; i += 32) {
    int c = i & (CDIM - 1);
    xl[i] = (__bf16)fmaxf(Y2[r0 * CDIM + i] * scale[c] + bias[c], 0.0f);
  }
  __syncthreads();

  const int hi = (lane & 16) ? 8 : 0;
  f32x8 acc[8];

  zero_acc(acc);
  gemm16x128x128(xl, s_w, lane, acc);
#pragma unroll
  for (int nb = 0; nb < 8; ++nb) {
    const int n = nb * 16 + (lane & 15);
#pragma unroll
    for (int j = 0; j < 8; ++j) Q[(r0 + hi + j) * CDIM + n] = acc[nb][j];
  }
  __syncthreads();
  stage_w(s_w, Wk, threadIdx.x, 128);
  __syncthreads();

  zero_acc(acc);
  gemm16x128x128(xl, s_w, lane, acc);
#pragma unroll
  for (int nb = 0; nb < 8; ++nb) {
    const int n = nb * 16 + (lane & 15);
#pragma unroll
    for (int j = 0; j < 8; ++j) Kf[(r0 + hi + j) * CDIM + n] = acc[nb][j];
  }
  __syncthreads();
  stage_w(s_w, Wv, threadIdx.x, 128);
  __syncthreads();

  zero_acc(acc);
  gemm16x128x128(xl, s_w, lane, acc);
#pragma unroll
  for (int nb = 0; nb < 8; ++nb) {
    const int n = nb * 16 + (lane & 15);
#pragma unroll
    for (int j = 0; j < 8; ++j) Vf[(r0 + hi + j) * CDIM + n] = acc[nb][j];
  }
}

// ---------------- fully fused per-point attention: one wave per point ----------------
// LDS (exactly 64 KB): 32 KB shared weight cache (Wd2 -> Wg1 -> Wg2)
//                    + per-wave: h/t (4 KB, reused), pos (4 KB), vv (8 KB)
__global__ void k_attn(const float* __restrict__ pose_enc, const int* __restrict__ knn,
                       const float* __restrict__ Q, const float* __restrict__ Kf,
                       const float* __restrict__ Vf,
                       const float* __restrict__ d_w1, const float* __restrict__ d_b1,
                       const __bf16* __restrict__ Wd2, const float* __restrict__ d_b2,
                       const __bf16* __restrict__ Wg1, const float* __restrict__ g_b1,
                       const __bf16* __restrict__ Wg2, const float* __restrict__ g_b2,
                       float* __restrict__ Y1) {
  __shared__ __bf16 s_w[CDIM * CDIM];     // 32 KB shared weight cache
  __shared__ __bf16 s_h[2][KNN * CDIM];   // h tile, then t tile (bf16 A operand)
  __shared__ __bf16 s_p[2][KNN * CDIM];   // pos1, then pos (bf16)
  __shared__ float  s_vv[2][KNN * CDIM];  // gathered v rows

  const int wave = threadIdx.x >> 5, lane = threadIdx.x & 31;
  const int p = blockIdx.x * 2 + wave;    // 0 .. 16383, grid sized exactly
  const int b = p >> 12;                  // N = 4096

  stage_w(s_w, Wd2, threadIdx.x, 64);
  const int myidx = (lane < KNN) ? knn[p * KNN + lane] : 0;

  // pos1 = relu(pose_enc @ d_w1^T + d_b1)  (16x4 @ 4x128, VALU)
  {
    const int m  = lane & 15;
    const int c0 = (lane & 16) ? 64 : 0;
    const float* pe = pose_enc + (p * KNN + m) * 4;
    const float p0 = pe[0], p1 = pe[1], p2 = pe[2], p3 = pe[3];
    for (int cc = 0; cc < 64; ++cc) {
      const int c = c0 + cc;
      const float* w = d_w1 + c * 4;
      float s = d_b1[c] + p0 * w[0] + p1 * w[1] + p2 * w[2] + p3 * w[3];
      s_p[wave][m * CDIM + c] = (__bf16)fmaxf(s, 0.0f);
    }
  }
  __syncthreads();

  // pos = pos1 @ d_w2^T  (bias added separately; pacc stays live for epilogue)
  f32x8 pacc[8]; zero_acc(pacc);
  gemm16x128x128(s_p[wave], s_w, lane, pacc);
  __syncthreads();                        // all reads of s_p / s_w done

  const int hi = (lane & 16) ? 8 : 0;
  float db2v[8];
#pragma unroll
  for (int nb = 0; nb < 8; ++nb) {
    const int n = nb * 16 + (lane & 15);
    db2v[nb] = d_b2[n];
#pragma unroll
    for (int j = 0; j < 8; ++j)
      s_p[wave][(hi + j) * CDIM + n] = (__bf16)(pacc[nb][j] + db2v[nb]);
  }
  stage_w(s_w, Wg1, threadIdx.x, 64);

  float qv[4];
#pragma unroll
  for (int i = 0; i < 4; ++i) qv[i] = Q[p * CDIM + lane * 4 + i];
  __syncthreads();

  // gather kk/vv rows, h = q - kk + pos
  for (int k = 0; k < KNN; ++k) {
    const int row = __shfl(myidx, k, 32);
    const float* kfr = Kf + ((size_t)(b * 4096 + row)) * CDIM;
    const float* vfr = Vf + ((size_t)(b * 4096 + row)) * CDIM;
#pragma unroll
    for (int i = 0; i < 4; ++i) {
      const int c = lane * 4 + i;
      const float kkv = kfr[c];
      s_vv[wave][k * CDIM + c] = vfr[c];
      const float hv = qv[i] - kkv + (float)s_p[wave][k * CDIM + c];
      s_h[wave][k * CDIM + c] = (__bf16)hv;
    }
  }
  __syncthreads();

  // t = relu(h @ g_w1^T + g_b1)
  f32x8 acc[8]; zero_acc(acc);
  gemm16x128x128(s_h[wave], s_w, lane, acc);
  __syncthreads();                        // all reads of s_h / s_w done

  // write t back into the h buffer; restage Wg2
#pragma unroll
  for (int nb = 0; nb < 8; ++nb) {
    const int n = nb * 16 + (lane & 15);
    const float bv = g_b1[n];
#pragma unroll
    for (int j = 0; j < 8; ++j)
      s_h[wave][(hi + j) * CDIM + n] = (__bf16)fmaxf(acc[nb][j] + bv, 0.0f);
  }
  stage_w(s_w, Wg2, threadIdx.x, 64);
  __syncthreads();

  // attn_pre = t @ g_w2^T + g_b2 ; softmax over K ; res = sum_k attn*(vv+pos)
  zero_acc(acc);
  gemm16x128x128(s_h[wave], s_w, lane, acc);

  const float invs = 0.08838834764831845f;  // 1/sqrt(128)
#pragma unroll
  for (int nb = 0; nb < 8; ++nb) {
    const int n = nb * 16 + (lane & 15);
    const float gb = g_b2[n];
    float v[8], mx = -1e30f;
#pragma unroll
    for (int j = 0; j < 8; ++j) { v[j] = (acc[nb][j] + gb) * invs; mx = fmaxf(mx, v[j]); }
    mx = fmaxf(mx, __shfl_xor(mx, 16, 32));           // combine lane-pair halves (rows 0-7 / 8-15)
    float s = 0.0f;
#pragma unroll
    for (int j = 0; j < 8; ++j) { v[j] = __expf(v[j] - mx); s += v[j]; }
    s += __shfl_xor(s, 16, 32);
    float r = 0.0f;
#pragma unroll
    for (int j = 0; j < 8; ++j) {
      const float vp = s_vv[wave][(hi + j) * CDIM + n] + pacc[nb][j] + db2v[nb];
      r += (v[j] / s) * vp;
    }
    r += __shfl_xor(r, 16, 32);
    if (!(lane & 16)) Y1[p * CDIM + n] = r;
  }
}

// ---------------- epilogue: out = relu(bn1(z)) + new_points ----------------
__global__ void k_final(const float* __restrict__ Z, const float* __restrict__ scale,
                        const float* __restrict__ bias, const float* __restrict__ NP,
                        float* __restrict__ out) {
  const int i = blockIdx.x * 256 + threadIdx.x;      // 2M, grid sized exactly
  const int c = i & (CDIM - 1);
  out[i] = fmaxf(Z[i] * scale[c] + bias[c], 0.0f) + NP[i];
}

// ---------------- host ----------------
extern "C" void kernel_launch(void* const* d_in, const int* in_sizes, int n_in,
                              void* d_out, int out_size, void* d_ws, size_t ws_size,
                              hipStream_t stream) {
  const float* pose_enc = (const float*)d_in[0];
  const int*   knn      = (const int*)  d_in[1];
  const float* np_      = (const float*)d_in[2];
  const float* wq   = (const float*)d_in[3];
  const float* wk   = (const float*)d_in[4];
  const float* wv   = (const float*)d_in[5];
  const float* g_w1 = (const float*)d_in[6];
  const float* g_b1 = (const float*)d_in[7];
  const float* g_w2 = (const float*)d_in[8];
  const float* g_b2 = (const float*)d_in[9];
  const float* d_w1 = (const float*)d_in[10];
  const float* d_b1 = (const float*)d_in[11];
  const float* d_b2 = (const float*)d_in[13];
  const float* c1_w = (const float*)d_in[14];
  const float* c1_b = (const float*)d_in[15];
  const float* bn1_g = (const float*)d_in[16];
  const float* bn1_b = (const float*)d_in[17];
  const float* c2_w = (const float*)d_in[18];
  const float* c2_b = (const float*)d_in[19];
  const float* bn2_g = (const float*)d_in[20];
  const float* bn2_b = (const float*)d_in[21];

  char* ws = (char*)d_ws;
  __bf16* WB = (__bf16*)ws;                       // 8 x 128x128 bf16 weights
  float* y2 = (float*)(ws + 8 * 16384 * 2);
  float* q  = y2 + NPTS * CDIM;
  float* kf = q  + NPTS * CDIM;
  float* vf = kf + NPTS * CDIM;
  float* y1 = vf + NPTS * CDIM;
  float* z  = y1 + NPTS * CDIM;
  float* st = z  + NPTS * CDIM;                   // 1024 floats of BN stats
  float* sum2 = st,       *ssq2 = st + 128, *sc2 = st + 256, *bi2 = st + 384;
  float* sum1 = st + 512, *ssq1 = st + 640, *sc1 = st + 768, *bi1 = st + 896;

  k_init<<<1, 256, 0, stream>>>(st);

  WPtrs wp;
  wp.src[0] = c2_w; wp.src[1] = wq;   wp.src[2] = wk;   wp.src[3] = wv;
  wp.src[4] = (const float*)d_in[12] /*d_w2*/; wp.src[5] = g_w1; wp.src[6] = g_w2; wp.src[7] = c1_w;
  k_cvt<<<512, 256, 0, stream>>>(wp, WB);

  // conv2 + BN2 stats
  k_conv_stats<<<256, 128, 0, stream>>>(np_, WB + 0 * 16384, c2_b, y2, sum2, ssq2);
  k_stats<<<1, 128, 0, stream>>>(sum2, ssq2, bn2_g, bn2_b, sc2, bi2);

  // q/k/v
  k_qkv<<<256, 128, 0, stream>>>(y2, sc2, bi2, WB + 1 * 16384, WB + 2 * 16384,
                                 WB + 3 * 16384, q, kf, vf);

  // fused per-point attention
  k_attn<<<NPTS / 2, 64, 0, stream>>>(pose_enc, knn, q, kf, vf,
                                      d_w1, d_b1, WB + 4 * 16384, d_b2,
                                      WB + 5 * 16384, g_b1, WB + 6 * 16384, g_b2, y1);

  // conv1 + BN1 stats
  k_conv_stats<<<256, 128, 0, stream>>>(y1, WB + 7 * 16384, c1_b, z, sum1, ssq1);
  k_stats<<<1, 128, 0, stream>>>(sum1, ssq1, bn1_g, bn1_b, sc1, bi1);

  k_final<<<NPTS * CDIM / 256, 256, 0, stream>>>(z, sc1, bi1, np_, (float*)d_out);
}
